// BLSTMGenerator_22720376996314
// MI455X (gfx1250) — compile-verified, matrix operand order checked by reading, not run on hardware
//
#include <hip/hip_runtime.h>
#include <stdint.h>

// ---------------------------------------------------------------------------
// Types for CDNA5 WMMA (wave32, 16x16x32 bf16 -> f32 accumulate)
// ---------------------------------------------------------------------------
typedef __bf16 bf16_t;
typedef __attribute__((ext_vector_type(16))) __bf16 v16bf;
typedef __attribute__((ext_vector_type(8)))  __bf16 v8bf;
typedef __attribute__((ext_vector_type(4)))  __bf16 v4bf;
typedef __attribute__((ext_vector_type(8)))  float  v8f;

#define TDIM 512
#define BDIM 128
#define EDIM 256
#define HDIM 512
#define VDIM 512
#define NGATE 2048   // 4*H

// ---------------------------------------------------------------------------
// Fragment helpers.
// ISA 16-bit A layout (16x32, MxK): lane = M (mod 16); lanes 0-15 hold
// K = {0..7, 16..23}, lanes 16-31 hold K = {8..15, 24..31}. Elements 0..7 of
// the v16bf are one contiguous 8-element K run, elements 8..15 the second.
// B (32x16, KxN) is symmetric with lane = N when the weight matrix is stored
// transposed (row = output unit, contiguous in K). Each run is 16 contiguous
// bytes -> global_load_b128.
// ---------------------------------------------------------------------------
__device__ __forceinline__ v16bf load_frag32(const bf16_t* p, int kb) {
  v8bf lo = *(const v8bf*)(p + kb);
  v8bf hi = *(const v8bf*)(p + 16 + kb);
  return __builtin_shufflevector(lo, hi, 0,1,2,3,4,5,6,7,8,9,10,11,12,13,14,15);
}

__device__ __forceinline__ v8f wmma_bf16(v16bf a, v16bf b, v8f c) {
  // (neg_a, A, neg_b, B, c_mod, C, reuse_a, reuse_b)
  return __builtin_amdgcn_wmma_f32_16x16x32_bf16(false, a, false, b, (short)0, c,
                                                 false, false);
}

__device__ __forceinline__ float sigf(float x) {
  return 1.0f / (1.0f + __expf(-x));
}
__device__ __forceinline__ float tanh_fast(float x) {
  return 1.0f - 2.0f / (__expf(2.0f * x) + 1.0f);
}

// ---------------------------------------------------------------------------
// Per-timestep gate GEMM: gates[B,4H] = [x_t ; h_{t-1}] @ [Wx;Wh] + bias
// Weights pre-transposed: Wt[n][k], k in [0, Kin+H).
// 2x2 register blocking: each wave computes a 32x32 output block
// (4 WMMA per k-step from 2 A-frags + 2 B-frags -> 2 b128 loads per WMMA).
// Grid: (32, ndir) blocks x 256 threads (8 waves).
// ---------------------------------------------------------------------------
struct GateArgs {
  const bf16_t* X;  int ldx; int Kin;
  const bf16_t* H;  int ldh;
  const bf16_t* Wt;           // [NGATE][Kin+HDIM]
  const float*  bias;         // [NGATE]
  float*        gates;        // [BDIM][NGATE]
};

__global__ __launch_bounds__(256) void lstm_gates_kernel(GateArgs a0, GateArgs a1) {
  const GateArgs a = (blockIdx.y == 0) ? a0 : a1;
  const int wave = threadIdx.x >> 5;
  const int lane = threadIdx.x & 31;
  const int grp = blockIdx.x * 8 + wave;   // 0..255
  const int mg = grp >> 6;                 // 0..3   (M = 128, 32 rows/group)
  const int ng = grp & 63;                 // 0..63  (N = 2048, 32 cols/group)
  const int r16 = lane & 15;
  const int kb  = (lane >> 4) << 3;        // 0 or 8
  const int m0 = (mg << 5) + r16;
  const int n0 = (ng << 5) + r16;
  const int ldw = a.Kin + HDIM;

  const bf16_t* arow0 = a.X + (size_t)m0 * a.ldx;
  const bf16_t* arow1 = arow0 + (size_t)16 * a.ldx;
  const bf16_t* hrow0 = a.H + (size_t)m0 * a.ldh;
  const bf16_t* hrow1 = hrow0 + (size_t)16 * a.ldh;
  const bf16_t* brow0 = a.Wt + (size_t)n0 * ldw;
  const bf16_t* brow1 = brow0 + (size_t)16 * ldw;

  v8f acc00 = {}, acc01 = {}, acc10 = {}, acc11 = {};
  for (int k0 = 0; k0 < a.Kin; k0 += 32) {
    v16bf A0 = load_frag32(arow0 + k0, kb);
    v16bf A1 = load_frag32(arow1 + k0, kb);
    v16bf B0 = load_frag32(brow0 + k0, kb);
    v16bf B1 = load_frag32(brow1 + k0, kb);
    acc00 = wmma_bf16(A0, B0, acc00);
    acc01 = wmma_bf16(A0, B1, acc01);
    acc10 = wmma_bf16(A1, B0, acc10);
    acc11 = wmma_bf16(A1, B1, acc11);
  }
  const bf16_t* wrow0 = brow0 + a.Kin;
  const bf16_t* wrow1 = brow1 + a.Kin;
#pragma unroll 2
  for (int k0 = 0; k0 < HDIM; k0 += 32) {
    v16bf A0 = load_frag32(hrow0 + k0, kb);
    v16bf A1 = load_frag32(hrow1 + k0, kb);
    v16bf B0 = load_frag32(wrow0 + k0, kb);
    v16bf B1 = load_frag32(wrow1 + k0, kb);
    acc00 = wmma_bf16(A0, B0, acc00);
    acc01 = wmma_bf16(A0, B1, acc01);
    acc10 = wmma_bf16(A1, B0, acc10);
    acc11 = wmma_bf16(A1, B1, acc11);
  }

  // C/D layout: VGPR r holds row (base + r + 8*(lane>=16)), col = lane&15.
  const float bn0 = a.bias[n0];
  const float bn1 = a.bias[n0 + 16];
  const int rb = (mg << 5) + ((lane >> 4) << 3);
  float* gp0 = a.gates + n0;
  float* gp1 = a.gates + n0 + 16;
#pragma unroll
  for (int r = 0; r < 8; ++r) {
    gp0[(size_t)(rb + r) * NGATE]      = acc00[r] + bn0;
    gp1[(size_t)(rb + r) * NGATE]      = acc01[r] + bn1;
    gp0[(size_t)(rb + 16 + r) * NGATE] = acc10[r] + bn0;
    gp1[(size_t)(rb + 16 + r) * NGATE] = acc11[r] + bn1;
  }
}

// ---------------------------------------------------------------------------
// LSTM pointwise: c' = sig(f)*c + sig(i)*tanh(g); h' = sig(o)*tanh(c')
// Grid: (B*H/256, ndir) x 256 threads.
// ---------------------------------------------------------------------------
struct PwArgs {
  const float* gates;          // [BDIM][NGATE]
  float*  c;  int ldc;         // c state (f32), possibly column-offset view
  bf16_t* h;  int ldh;         // h state (bf16) for next-step GEMM
  bf16_t* hs; int ldhs;        // optional per-step output (pre-offset), or null
  bf16_t* cbf;                 // optional bf16 copy of c (for bridge), or null
};

__global__ __launch_bounds__(256) void lstm_pointwise_kernel(PwArgs p0, PwArgs p1) {
  const PwArgs p = (blockIdx.y == 0) ? p0 : p1;
  const int idx = blockIdx.x * 256 + threadIdx.x;  // 0 .. B*H-1
  const int b = idx >> 9;
  const int j = idx & (HDIM - 1);
  const float* g = p.gates + (size_t)b * NGATE;
  const float ig = g[j];
  const float fg = g[j + HDIM];
  const float gg = g[j + 2 * HDIM];
  const float og = g[j + 3 * HDIM];
  const float cold = p.c[(size_t)b * p.ldc + j];
  const float cn = sigf(fg) * cold + sigf(ig) * tanh_fast(gg);
  const float hn = sigf(og) * tanh_fast(cn);
  p.c[(size_t)b * p.ldc + j] = cn;
  p.h[(size_t)b * p.ldh + j] = (bf16_t)hn;
  if (p.hs)  p.hs[(size_t)b * p.ldhs + j] = (bf16_t)hn;
  if (p.cbf) p.cbf[(size_t)b * p.ldc + j] = (bf16_t)cn;
}

// ---------------------------------------------------------------------------
// Generic GEMM: out[M,N] = A[M,K](bf16) * Bt[N,K]^T + bias, optional tanh,
// optional [T,B]->[B,T] permuted store (for final logits).
// 2x2 register blocking, wave per 32x32 block, 8 waves/block.
// Requires M % 32 == 0, N % 32 == 0, K % 32 == 0.
// ---------------------------------------------------------------------------
__global__ __launch_bounds__(256) void gemm_bt_kernel(
    const bf16_t* A, int lda, const bf16_t* Bt, int ldb, const float* bias,
    float* out, int M, int N, int K, int act_tanh, int permTB) {
  const int wave = threadIdx.x >> 5;
  const int lane = threadIdx.x & 31;
  const int Ng = N >> 5;
  const int grp = blockIdx.x * 8 + wave;
  const int total = (M >> 5) * Ng;
  if (grp >= total) return;
  const int mg = grp / Ng;
  const int ng = grp - mg * Ng;
  const int r16 = lane & 15;
  const int kb = (lane >> 4) << 3;
  const int m0 = (mg << 5) + r16;
  const int n0 = (ng << 5) + r16;

  const bf16_t* arow0 = A + (size_t)m0 * lda;
  const bf16_t* arow1 = arow0 + (size_t)16 * lda;
  const bf16_t* brow0 = Bt + (size_t)n0 * ldb;
  const bf16_t* brow1 = brow0 + (size_t)16 * ldb;

  v8f acc00 = {}, acc01 = {}, acc10 = {}, acc11 = {};
#pragma unroll 2
  for (int k0 = 0; k0 < K; k0 += 32) {
    v16bf A0 = load_frag32(arow0 + k0, kb);
    v16bf A1 = load_frag32(arow1 + k0, kb);
    v16bf B0 = load_frag32(brow0 + k0, kb);
    v16bf B1 = load_frag32(brow1 + k0, kb);
    acc00 = wmma_bf16(A0, B0, acc00);
    acc01 = wmma_bf16(A0, B1, acc01);
    acc10 = wmma_bf16(A1, B0, acc10);
    acc11 = wmma_bf16(A1, B1, acc11);
  }

  const float bn0 = bias ? bias[n0] : 0.0f;
  const float bn1 = bias ? bias[n0 + 16] : 0.0f;
  const int rb = (mg << 5) + ((lane >> 4) << 3);
#pragma unroll
  for (int r = 0; r < 8; ++r) {
#pragma unroll
    for (int i = 0; i < 2; ++i) {
      const int row = rb + 16 * i + r;
      float v0 = (i == 0 ? acc00[r] : acc10[r]) + bn0;
      float v1 = (i == 0 ? acc01[r] : acc11[r]) + bn1;
      if (act_tanh) { v0 = tanh_fast(v0); v1 = tanh_fast(v1); }
      size_t off;
      if (permTB) {  // rows of A ordered [t][b]; output is [b][t][n]
        const int t = row / BDIM;
        const int b = row - t * BDIM;
        off = ((size_t)b * TDIM + t) * (size_t)N + n0;
      } else {
        off = (size_t)row * N + n0;
      }
      out[off] = v0;
      out[off + 16] = v1;
    }
  }
}

// ---------------------------------------------------------------------------
// Setup kernels: embedding gather (f32 -> bf16, [B,T] tokens -> [T,B,E]) and
// weight conversion to transposed bf16 (Wt[n][k], [Wx;Wh] fused along K).
// ---------------------------------------------------------------------------
__global__ void embed_kernel(const int* x, const float* emb, bf16_t* out) {
  const size_t total = (size_t)TDIM * BDIM * (EDIM / 4);
  for (size_t i = (size_t)blockIdx.x * blockDim.x + threadIdx.x; i < total;
       i += (size_t)gridDim.x * blockDim.x) {
    const int e4 = (int)(i % (EDIM / 4));
    const size_t tb = i / (EDIM / 4);
    const int b = (int)(tb % BDIM);
    const int t = (int)(tb / BDIM);
    const int tok = x[(size_t)b * TDIM + t];
    const float4 v = *(const float4*)(emb + (size_t)tok * EDIM + e4 * 4);
    v4bf o;
    o[0] = (bf16_t)v.x; o[1] = (bf16_t)v.y; o[2] = (bf16_t)v.z; o[3] = (bf16_t)v.w;
    *(v4bf*)(out + ((size_t)t * BDIM + b) * EDIM + e4 * 4) = o;
  }
}

__global__ void convert_wt_kernel(const float* Wx, const float* Wh, bf16_t* Wt,
                                  int Kin, int Kh, int N) {
  const int ld = Kin + Kh;
  const size_t total = (size_t)N * ld;
  for (size_t i = (size_t)blockIdx.x * blockDim.x + threadIdx.x; i < total;
       i += (size_t)gridDim.x * blockDim.x) {
    const int n = (int)(i / ld);
    const int k = (int)(i - (size_t)n * ld);
    const float v = (k < Kin) ? Wx[(size_t)k * N + n] : Wh[(size_t)(k - Kin) * N + n];
    Wt[i] = (bf16_t)v;
  }
}

__global__ void transpose_bf16_kernel(const float* W, bf16_t* Wt, int K, int N) {
  const size_t total = (size_t)N * K;
  for (size_t i = (size_t)blockIdx.x * blockDim.x + threadIdx.x; i < total;
       i += (size_t)gridDim.x * blockDim.x) {
    const int n = (int)(i / K);
    const int k = (int)(i - (size_t)n * K);
    Wt[i] = (bf16_t)W[(size_t)k * N + n];
  }
}

__global__ void dec_init_kernel(const float* hb, const float* cb, bf16_t* h0,
                                bf16_t* h1, float* c0, float* c1) {
  const int i = blockIdx.x * 256 + threadIdx.x;  // B*H
  const float hv = hb[i];
  const float cv = cb[i];
  h0[i] = (bf16_t)hv;
  h1[i] = (bf16_t)hv;
  c0[i] = cv;
  c1[i] = cv;
}

// ---------------------------------------------------------------------------
// Host orchestration
// ---------------------------------------------------------------------------
extern "C" void kernel_launch(void* const* d_in, const int* in_sizes, int n_in,
                              void* d_out, int out_size, void* d_ws, size_t ws_size,
                              hipStream_t stream) {
  (void)in_sizes; (void)n_in; (void)out_size; (void)ws_size;

  const int*   x         = (const int*)d_in[0];
  const float* emb       = (const float*)d_in[1];
  const float* encfWx0   = (const float*)d_in[2];
  const float* encfWh0   = (const float*)d_in[3];
  const float* encfb0    = (const float*)d_in[4];
  const float* encfWx1   = (const float*)d_in[5];
  const float* encfWh1   = (const float*)d_in[6];
  const float* encfb1    = (const float*)d_in[7];
  const float* encbWx0   = (const float*)d_in[8];
  const float* encbWh0   = (const float*)d_in[9];
  const float* encbb0    = (const float*)d_in[10];
  const float* encbWx1   = (const float*)d_in[11];
  const float* encbWh1   = (const float*)d_in[12];
  const float* encbb1    = (const float*)d_in[13];
  const float* decWx0    = (const float*)d_in[14];
  const float* decWh0    = (const float*)d_in[15];
  const float* decb0     = (const float*)d_in[16];
  const float* decWx1    = (const float*)d_in[17];
  const float* decWh1    = (const float*)d_in[18];
  const float* decb1     = (const float*)d_in[19];
  const float* hprojW    = (const float*)d_in[20];
  const float* hprojb    = (const float*)d_in[21];
  const float* cprojW    = (const float*)d_in[22];
  const float* cprojb    = (const float*)d_in[23];
  const float* fcW       = (const float*)d_in[24];
  const float* fcb       = (const float*)d_in[25];

  char* ws = (char*)d_ws;
  size_t off = 0;
  auto alloc = [&](size_t bytes) -> void* {
    off = (off + 255) & ~((size_t)255);
    void* p = ws + off;
    off += bytes;
    return p;
  };

  // Activations
  bf16_t* embT   = (bf16_t*)alloc((size_t)TDIM * BDIM * EDIM * 2);     // [T,B,E]
  bf16_t* hs0    = (bf16_t*)alloc((size_t)TDIM * BDIM * 1024 * 2);     // [T,B,2H]
  bf16_t* dec_hs = (bf16_t*)alloc((size_t)TDIM * BDIM * HDIM * 2);     // [T,B,H]
  // Transposed bf16 weights  Wt[n][k]
  bf16_t* wEncF0 = (bf16_t*)alloc((size_t)NGATE * 768 * 2);
  bf16_t* wEncB0 = (bf16_t*)alloc((size_t)NGATE * 768 * 2);
  bf16_t* wEncF1 = (bf16_t*)alloc((size_t)NGATE * 1536 * 2);
  bf16_t* wEncB1 = (bf16_t*)alloc((size_t)NGATE * 1536 * 2);
  bf16_t* wDec0  = (bf16_t*)alloc((size_t)NGATE * 768 * 2);
  bf16_t* wDec1  = (bf16_t*)alloc((size_t)NGATE * 1024 * 2);
  bf16_t* hprojT = (bf16_t*)alloc((size_t)HDIM * 1024 * 2);
  bf16_t* cprojT = (bf16_t*)alloc((size_t)HDIM * 1024 * 2);
  bf16_t* fcT    = (bf16_t*)alloc((size_t)VDIM * HDIM * 2);
  // Recurrent state
  bf16_t* h_f0   = (bf16_t*)alloc((size_t)BDIM * HDIM * 2);
  bf16_t* h_b0   = (bf16_t*)alloc((size_t)BDIM * HDIM * 2);
  float*  c_f0   = (float*) alloc((size_t)BDIM * HDIM * 4);
  float*  c_b0   = (float*) alloc((size_t)BDIM * HDIM * 4);
  bf16_t* hcat   = (bf16_t*)alloc((size_t)BDIM * 1024 * 2);  // enc L1 [h_f|h_b]
  float*  ccat   = (float*) alloc((size_t)BDIM * 1024 * 4);
  bf16_t* cbfcat = (bf16_t*)alloc((size_t)BDIM * 1024 * 2);
  bf16_t* dh0    = (bf16_t*)alloc((size_t)BDIM * HDIM * 2);
  bf16_t* dh1    = (bf16_t*)alloc((size_t)BDIM * HDIM * 2);
  float*  dc0    = (float*) alloc((size_t)BDIM * HDIM * 4);
  float*  dc1    = (float*) alloc((size_t)BDIM * HDIM * 4);
  float*  g0     = (float*) alloc((size_t)BDIM * NGATE * 4);
  float*  g1     = (float*) alloc((size_t)BDIM * NGATE * 4);
  float*  hbridge= (float*) alloc((size_t)BDIM * HDIM * 4);
  float*  cbridge= (float*) alloc((size_t)BDIM * HDIM * 4);

  // ---- zero initial encoder state ----
  hipMemsetAsync(h_f0, 0, (size_t)BDIM * HDIM * 2, stream);
  hipMemsetAsync(h_b0, 0, (size_t)BDIM * HDIM * 2, stream);
  hipMemsetAsync(c_f0, 0, (size_t)BDIM * HDIM * 4, stream);
  hipMemsetAsync(c_b0, 0, (size_t)BDIM * HDIM * 4, stream);
  hipMemsetAsync(hcat, 0, (size_t)BDIM * 1024 * 2, stream);
  hipMemsetAsync(ccat, 0, (size_t)BDIM * 1024 * 4, stream);

  // ---- weight conversion + embedding ----
  convert_wt_kernel<<<2048, 256, 0, stream>>>(encfWx0, encfWh0, wEncF0, EDIM, HDIM, NGATE);
  convert_wt_kernel<<<2048, 256, 0, stream>>>(encbWx0, encbWh0, wEncB0, EDIM, HDIM, NGATE);
  convert_wt_kernel<<<4096, 256, 0, stream>>>(encfWx1, encfWh1, wEncF1, 1024, HDIM, NGATE);
  convert_wt_kernel<<<4096, 256, 0, stream>>>(encbWx1, encbWh1, wEncB1, 1024, HDIM, NGATE);
  convert_wt_kernel<<<2048, 256, 0, stream>>>(decWx0, decWh0, wDec0, EDIM, HDIM, NGATE);
  convert_wt_kernel<<<2048, 256, 0, stream>>>(decWx1, decWh1, wDec1, HDIM, HDIM, NGATE);
  transpose_bf16_kernel<<<1024, 256, 0, stream>>>(hprojW, hprojT, 1024, HDIM);
  transpose_bf16_kernel<<<1024, 256, 0, stream>>>(cprojW, cprojT, 1024, HDIM);
  transpose_bf16_kernel<<<512, 256, 0, stream>>>(fcW, fcT, HDIM, VDIM);
  embed_kernel<<<8192, 256, 0, stream>>>(x, emb, embT);

  const dim3 blk(256);
  const dim3 grdGemm2(32, 2), grdGemm1(32, 1);
  const dim3 grdPw2(256, 2), grdPw1(256, 1);

  // ---- encoder layer 0 (fwd + bwd fused over grid.y) ----
  for (int t = 0; t < TDIM; ++t) {
    const int tb = TDIM - 1 - t;
    GateArgs af{embT + (size_t)t * BDIM * EDIM, EDIM, EDIM, h_f0, HDIM,
                wEncF0, encfb0, g0};
    GateArgs ab{embT + (size_t)tb * BDIM * EDIM, EDIM, EDIM, h_b0, HDIM,
                wEncB0, encbb0, g1};
    lstm_gates_kernel<<<grdGemm2, blk, 0, stream>>>(af, ab);
    PwArgs pf{g0, c_f0, HDIM, h_f0, HDIM, hs0 + (size_t)t * BDIM * 1024, 1024, nullptr};
    PwArgs pb{g1, c_b0, HDIM, h_b0, HDIM, hs0 + (size_t)tb * BDIM * 1024 + HDIM, 1024, nullptr};
    lstm_pointwise_kernel<<<grdPw2, blk, 0, stream>>>(pf, pb);
  }

  // ---- encoder layer 1 (input = [hs_f|hs_b], states in concat buffers) ----
  for (int t = 0; t < TDIM; ++t) {
    const int tb = TDIM - 1 - t;
    GateArgs af{hs0 + (size_t)t * BDIM * 1024, 1024, 1024, hcat, 1024,
                wEncF1, encfb1, g0};
    GateArgs ab{hs0 + (size_t)tb * BDIM * 1024, 1024, 1024, hcat + HDIM, 1024,
                wEncB1, encbb1, g1};
    lstm_gates_kernel<<<grdGemm2, blk, 0, stream>>>(af, ab);
    PwArgs pf{g0, ccat, 1024, hcat, 1024, nullptr, 0, cbfcat};
    PwArgs pb{g1, ccat + HDIM, 1024, hcat + HDIM, 1024, nullptr, 0, cbfcat + HDIM};
    lstm_pointwise_kernel<<<grdPw2, blk, 0, stream>>>(pf, pb);
  }

  // ---- bridge: dec_h = tanh(h_enc @ hproj + b), dec_c = tanh(c_enc @ cproj + b)
  gemm_bt_kernel<<<8, blk, 0, stream>>>(hcat, 1024, hprojT, 1024, hprojb,
                                        hbridge, BDIM, HDIM, 1024, 1, 0);
  gemm_bt_kernel<<<8, blk, 0, stream>>>(cbfcat, 1024, cprojT, 1024, cprojb,
                                        cbridge, BDIM, HDIM, 1024, 1, 0);
  dec_init_kernel<<<256, blk, 0, stream>>>(hbridge, cbridge, dh0, dh1, dc0, dc1);

  // ---- decoder: 2-layer uni-LSTM ----
  for (int t = 0; t < TDIM; ++t) {
    GateArgs a0{embT + (size_t)t * BDIM * EDIM, EDIM, EDIM, dh0, HDIM,
                wDec0, decb0, g0};
    lstm_gates_kernel<<<grdGemm1, blk, 0, stream>>>(a0, a0);
    PwArgs p0{g0, dc0, HDIM, dh0, HDIM, nullptr, 0, nullptr};
    lstm_pointwise_kernel<<<grdPw1, blk, 0, stream>>>(p0, p0);

    GateArgs a1{dh0, HDIM, HDIM, dh1, HDIM, wDec1, decb1, g1};
    lstm_gates_kernel<<<grdGemm1, blk, 0, stream>>>(a1, a1);
    PwArgs p1{g1, dc1, HDIM, dh1, HDIM, dec_hs + (size_t)t * BDIM * HDIM, HDIM, nullptr};
    lstm_pointwise_kernel<<<grdPw1, blk, 0, stream>>>(p1, p1);
  }

  // ---- final FC: logits[B,T,V] = dec_hs[T,B,H] @ fc_W + fc_b (permuted store)
  const int fcGroups = ((TDIM * BDIM) / 32) * (VDIM / 32);  // 32768
  gemm_bt_kernel<<<fcGroups / 8, blk, 0, stream>>>(
      dec_hs, HDIM, fcT, HDIM, fcb, (float*)d_out, TDIM * BDIM, VDIM, HDIM, 0, 1);
}
